// GNNModel_72387378807366
// MI455X (gfx1250) — matrix operand order repaired and checked by reading, not compiled, not run on hardware
//
#include <hip/hip_runtime.h>
#include <hip/hip_bf16.h>
#include <hip/hip_fp16.h>

typedef __attribute__((ext_vector_type(16))) _Float16 v16h;
typedef __attribute__((ext_vector_type(8)))  float    v8f;

#define NEG_SLOPE 0.2f

__device__ __forceinline__ float lrelu(float x) { return x > 0.f ? x : NEG_SLOPE * x; }
// monotone float <-> uint encoding for atomicMax on floats (handles negatives)
__device__ __forceinline__ unsigned fenc(float f) {
    unsigned u = __float_as_uint(f);
    return (u & 0x80000000u) ? ~u : (u | 0x80000000u);
}
__device__ __forceinline__ float fdec(unsigned u) {
    return __uint_as_float((u & 0x80000000u) ? (u & 0x7FFFFFFFu) : ~u);
}

// ---------------------------------------------------------------------------
// GEMM: Hout[M,64] = (relu_in ? relu(A) : A)[M,K] @ W[K,64], via v_wmma_f32_16x16x32_f16
// One wave per 16-row M-tile; 4 N-tiles (64 cols) per wave; K in 32-chunks.
// ---------------------------------------------------------------------------
__global__ __launch_bounds__(256) void k_gemm_wmma(
    const float* __restrict__ A, const float* __restrict__ W,
    float* __restrict__ Hout, int M, int K, int relu_in)
{
    const int lane  = threadIdx.x & 31;
    const int mtile = blockIdx.x * (blockDim.x >> 5) + (threadIdx.x >> 5);
    if (mtile * 16 >= M) return;              // whole-wave uniform exit (EXEC stays all-ones)
    const int mbase = mtile * 16;
    const int halfA = (lane < 16) ? 0 : 8;    // A K-offset selector within 32-block
    const int halfB = (lane < 16) ? 0 : 8;    // B K-offset selector within 32-block
    const int rowA  = mbase + (lane & 15);
    const int colL  = lane & 15;

    v8f acc[4];
    #pragma unroll
    for (int t = 0; t < 4; ++t)
        #pragma unroll
        for (int i = 0; i < 8; ++i) acc[t][i] = 0.f;

    for (int kk = 0; kk < K; kk += 32) {
        // A fragment: lane<16 -> K {kk+0..7, kk+16..23}; lane>=16 -> {kk+8..15, kk+24..31}
        const float* ap = A + (size_t)rowA * K + kk + halfA;
        float4 f0 = ((const float4*)ap)[0];
        float4 f1 = ((const float4*)ap)[1];
        float4 f2 = ((const float4*)(ap + 16))[0];
        float4 f3 = ((const float4*)(ap + 16))[1];
        float fa[16] = { f0.x, f0.y, f0.z, f0.w, f1.x, f1.y, f1.z, f1.w,
                         f2.x, f2.y, f2.z, f2.w, f3.x, f3.y, f3.z, f3.w };
        v16h a;
        #pragma unroll
        for (int i = 0; i < 16; ++i) {
            float v = fa[i];
            if (relu_in) v = fmaxf(v, 0.f);
            a[i] = (_Float16)v;
        }
        #pragma unroll
        for (int t = 0; t < 4; ++t) {
            // B fragment: lane holds column (t*16 + colL), K {kb+0..7, kb+16..23}
            const int col = t * 16 + colL;
            const int kb  = kk + halfB;
            v16h b;
            #pragma unroll
            for (int i = 0; i < 16; ++i)
                b[i] = (_Float16)W[(size_t)(kb + i + ((i >> 3) << 3)) * 64 + col];
            acc[t] = __builtin_amdgcn_wmma_f32_16x16x32_f16(
                false, a, false, b, (short)0, acc[t], false, false);
        }
    }
    // D layout: VGPR r -> row mbase + r (+8 for lanes 16-31), col = tile + lane%16
    const int r0 = (lane < 16) ? 0 : 8;
    #pragma unroll
    for (int t = 0; t < 4; ++t)
        #pragma unroll
        for (int r = 0; r < 8; ++r)
            Hout[(size_t)(mbase + r0 + r) * 64 + t * 16 + colL] = acc[t][r];
}

// ---------------------------------------------------------------------------
// Small/setup kernels
// ---------------------------------------------------------------------------
__global__ void k_zero_small(float* sums) {
    if (threadIdx.x < 2) sums[threadIdx.x] = 0.f;
}

__global__ __launch_bounds__(256) void k_edge_attr_sum(
    const float* __restrict__ ea, float* __restrict__ sums, int E)
{
    __shared__ float s0[256], s1[256];
    float a0 = 0.f, a1 = 0.f;
    for (int i = blockIdx.x * blockDim.x + threadIdx.x; i < E; i += gridDim.x * blockDim.x) {
        a0 += ea[2 * i];
        a1 += ea[2 * i + 1];
    }
    s0[threadIdx.x] = a0; s1[threadIdx.x] = a1;
    __syncthreads();
    for (int s = 128; s > 0; s >>= 1) {
        if (threadIdx.x < s) { s0[threadIdx.x] += s0[threadIdx.x + s]; s1[threadIdx.x] += s1[threadIdx.x + s]; }
        __syncthreads();
    }
    if (threadIdx.x == 0) { atomicAdd(&sums[0], s0[0]); atomicAdd(&sums[1], s1[0]); }
}

// c = We @ ae (2 floats); aeself = mean(edge_attr) . c  -> cbuf = {c0, c1, aeself}
__global__ void k_compute_c(const float* __restrict__ We, const float* __restrict__ ae,
                            const float* __restrict__ sums, float* __restrict__ cbuf, float invE)
{
    __shared__ float r0[64], r1[64];
    int t = threadIdx.x;
    r0[t] = We[t] * ae[t];
    r1[t] = We[64 + t] * ae[t];
    __syncthreads();
    for (int s = 32; s > 0; s >>= 1) {
        if (t < s) { r0[t] += r0[t + s]; r1[t] += r1[t + s]; }
        __syncthreads();
    }
    if (t == 0) {
        cbuf[0] = r0[0];
        cbuf[1] = r1[0];
        cbuf[2] = sums[0] * invE * r0[0] + sums[1] * invE * r1[0];
    }
}

// a_src[n] = h[n].as ; a_dst[n] = h[n].ad
__global__ __launch_bounds__(256) void k_node_attn(
    const float* __restrict__ Hf, const float* __restrict__ as_,
    const float* __restrict__ ad_, float* a_src, float* a_dst, int Nn)
{
    int n = blockIdx.x * blockDim.x + threadIdx.x;
    if (n >= Nn) return;
    const float4* hp = (const float4*)(Hf + (size_t)n * 64);
    const float4* ap = (const float4*)as_;
    const float4* dp = (const float4*)ad_;
    float s = 0.f, d = 0.f;
    #pragma unroll
    for (int i = 0; i < 16; ++i) {
        float4 h4 = hp[i], a4 = ap[i], d4 = dp[i];
        s += h4.x * a4.x + h4.y * a4.y + h4.z * a4.z + h4.w * a4.w;
        d += h4.x * d4.x + h4.y * d4.y + h4.z * d4.z + h4.w * d4.w;
    }
    a_src[n] = s; a_dst[n] = d;
}

// self-loop alpha per node; seeds segment-max
__global__ __launch_bounds__(256) void k_node_init(
    const float* __restrict__ a_src, const float* __restrict__ a_dst,
    const float* __restrict__ cbuf, float* alpha_self, unsigned* amax, int Nn)
{
    int n = blockIdx.x * blockDim.x + threadIdx.x;
    if (n >= Nn) return;
    float a = lrelu(a_src[n] + a_dst[n] + cbuf[2]);
    alpha_self[n] = a;
    amax[n] = fenc(a);
}

// edge pass 1: alpha + atomic segment max
__global__ __launch_bounds__(256) void k_edge_alpha(
    const int* __restrict__ src, const int* __restrict__ dst,
    const float* __restrict__ ea, const float* __restrict__ a_src,
    const float* __restrict__ a_dst, const float* __restrict__ cbuf,
    float* __restrict__ ebuf, unsigned* __restrict__ amax, int E)
{
    int e = blockIdx.x * blockDim.x + threadIdx.x;
    if (e >= E) return;
    int s = src[e], d = dst[e];
    float a = lrelu(a_src[s] + a_dst[d] + ea[2 * e] * cbuf[0] + ea[2 * e + 1] * cbuf[1]);
    ebuf[e] = a;
    atomicMax(&amax[d], fenc(a));
}

// decode amax in place; denom seeded with self-loop exp term
__global__ __launch_bounds__(256) void k_node_denom(
    const float* __restrict__ alpha_self, float* __restrict__ amax,
    float* __restrict__ denom, int Nn)
{
    int n = blockIdx.x * blockDim.x + threadIdx.x;
    if (n >= Nn) return;
    float m = fdec(((const unsigned*)amax)[n]);
    amax[n] = m;
    denom[n] = __expf(alpha_self[n] - m);
}

// edge pass 2: exp + atomic segment sum
__global__ __launch_bounds__(256) void k_edge_exp(
    const int* __restrict__ dst, float* __restrict__ ebuf,
    const float* __restrict__ amax, float* __restrict__ denom, int E)
{
    int e = blockIdx.x * blockDim.x + threadIdx.x;
    if (e >= E) return;
    int d = dst[e];
    float ex = __expf(ebuf[e] - amax[d]);
    ebuf[e] = ex;
    atomicAdd(&denom[d], ex);
}

// agg[n,:] = bias + h[n,:] * w_self    (initializes output; also carries bias)
__global__ __launch_bounds__(256) void k_agg_init(
    const float* __restrict__ Hf, const float* __restrict__ bias,
    const float* __restrict__ alpha_self, const float* __restrict__ amax,
    const float* __restrict__ denom, float* __restrict__ agg, int Nn)
{
    int i = blockIdx.x * blockDim.x + threadIdx.x;
    if (i >= Nn * 64) return;
    int n = i >> 6;
    float w = __expf(alpha_self[n] - amax[n]) / denom[n];
    agg[i] = bias[i & 63] + Hf[i] * w;
}

// edge pass 3: weighted scatter, one wave per edge; lane handles features lane & lane+32
__global__ __launch_bounds__(256) void k_edge_agg(
    const int* __restrict__ src, const int* __restrict__ dst,
    const float* __restrict__ ebuf, const float* __restrict__ denom,
    const float* __restrict__ Hf, float* __restrict__ agg, int E)
{
    int wid  = blockIdx.x * (blockDim.x >> 5) + (threadIdx.x >> 5);
    if (wid >= E) return;
    int lane = threadIdx.x & 31;
    int s = src[wid], d = dst[wid];
    float w = ebuf[wid] / denom[d];
    const float* hs = Hf + (size_t)s * 64;
    float* od = agg + (size_t)d * 64;
    atomicAdd(&od[lane],      hs[lane]      * w);
    atomicAdd(&od[lane + 32], hs[lane + 32] * w);
}

// head: out[n] = relu(agg2[n] . Wl + bl)
__global__ __launch_bounds__(256) void k_head(
    const float* __restrict__ agg, const float* __restrict__ Wl,
    const float* __restrict__ bl, float* __restrict__ out, int Nn)
{
    int n = blockIdx.x * blockDim.x + threadIdx.x;
    if (n >= Nn) return;
    const float4* hp = (const float4*)(agg + (size_t)n * 64);
    const float4* wp = (const float4*)Wl;
    float s = 0.f;
    #pragma unroll
    for (int i = 0; i < 16; ++i) {
        float4 h4 = hp[i], w4 = wp[i];
        s += h4.x * w4.x + h4.y * w4.y + h4.z * w4.z + h4.w * w4.w;
    }
    out[n] = fmaxf(s + bl[0], 0.f);
}

// ---------------------------------------------------------------------------
// One full GATConv layer
// ---------------------------------------------------------------------------
static void launch_gat_layer(
    const float* xin, int K, int relu_in,
    const float* W, const float* as_, const float* ad_,
    const float* We, const float* ae, const float* bias,
    const int* src, const int* dst, const float* ea,
    float* h, float* agg, float* a_src, float* a_dst, float* alpha_self,
    float* amax, float* denom, float* ebuf, float* sums, float* cbuf,
    int N, int E, hipStream_t stream)
{
    const int NB = (N + 255) / 256;
    const int EB = (E + 255) / 256;

    k_compute_c<<<1, 64, 0, stream>>>(We, ae, sums, cbuf, 1.f / (float)E);
    {
        int mtiles = N / 16;
        int blocks = (mtiles + 7) / 8;  // 8 waves (M-tiles) per 256-thread block
        k_gemm_wmma<<<blocks, 256, 0, stream>>>(xin, W, h, N, K, relu_in);
    }
    k_node_attn<<<NB, 256, 0, stream>>>(h, as_, ad_, a_src, a_dst, N);
    k_node_init<<<NB, 256, 0, stream>>>(a_src, a_dst, cbuf, alpha_self, (unsigned*)amax, N);
    k_edge_alpha<<<EB, 256, 0, stream>>>(src, dst, ea, a_src, a_dst, cbuf, ebuf, (unsigned*)amax, E);
    k_node_denom<<<NB, 256, 0, stream>>>(alpha_self, amax, denom, N);
    k_edge_exp<<<EB, 256, 0, stream>>>(dst, ebuf, amax, denom, E);
    k_agg_init<<<(N * 64 + 255) / 256, 256, 0, stream>>>(h, bias, alpha_self, amax, denom, agg, N);
    k_edge_agg<<<(E + 7) / 8, 256, 0, stream>>>(src, dst, ebuf, denom, h, agg, E);
}

extern "C" void kernel_launch(void* const* d_in, const int* in_sizes, int n_in,
                              void* d_out, int out_size, void* d_ws, size_t ws_size,
                              hipStream_t stream)
{
    const float* x   = (const float*)d_in[0];
    const int*   ei  = (const int*)d_in[1];
    const float* ea  = (const float*)d_in[2];
    const float* W1  = (const float*)d_in[3];
    const float* as1 = (const float*)d_in[4];
    const float* ad1 = (const float*)d_in[5];
    const float* We1 = (const float*)d_in[6];
    const float* ae1 = (const float*)d_in[7];
    const float* b1  = (const float*)d_in[8];
    const float* W2  = (const float*)d_in[9];
    const float* as2 = (const float*)d_in[10];
    const float* ad2 = (const float*)d_in[11];
    const float* We2 = (const float*)d_in[12];
    const float* ae2 = (const float*)d_in[13];
    const float* b2  = (const float*)d_in[14];
    const float* Wl  = (const float*)d_in[15];
    const float* bl  = (const float*)d_in[16];

    const int N = in_sizes[0] / 32;   // F_IN = 32
    const int E = in_sizes[1] / 2;
    const int* src = ei;
    const int* dst = ei + E;

    // workspace partition (floats)
    float* ws = (float*)d_ws;
    const size_t fH = (size_t)N * 64;
    float* h          = ws;
    float* agg        = h + fH;
    float* a_src      = agg + fH;
    float* a_dst      = a_src + N;
    float* alpha_self = a_dst + N;
    float* amax       = alpha_self + N;
    float* denom      = amax + N;
    float* ebuf       = denom + N;
    float* sums       = ebuf + E;
    float* cbuf       = sums + 2;
    (void)n_in; (void)out_size; (void)ws_size;

    // mean(edge_attr): zero + reduce (re-zeroed every call for replay determinism)
    k_zero_small<<<1, 32, 0, stream>>>(sums);
    k_edge_attr_sum<<<1024, 256, 0, stream>>>(ea, sums, E);

    // layer 1: input x (K=32), no input relu
    launch_gat_layer(x, 32, 0, W1, as1, ad1, We1, ae1, b1,
                     src, dst, ea, h, agg, a_src, a_dst, alpha_self,
                     amax, denom, ebuf, sums, cbuf, N, E, stream);

    // layer 2: input relu(agg1) (K=64); agg buffer is reused for agg2 after
    // it is no longer needed as GEMM/attention input
    launch_gat_layer(agg, 64, 1, W2, as2, ad2, We2, ae2, b2,
                     src, dst, ea, h, agg, a_src, a_dst, alpha_self,
                     amax, denom, ebuf, sums, cbuf, N, E, stream);

    // linear head + relu
    k_head<<<(N + 255) / 256, 256, 0, stream>>>(agg, Wl, bl, (float*)d_out, N);
}
